// MultimodalTopKSAE_80393197846569
// MI455X (gfx1250) — matrix operand order repaired and checked by reading, not compiled, not run on hardware
//
#include <hip/hip_runtime.h>
#include <math.h>

// ---------------------------------------------------------------------------
// MultimodalTopKSAE for MI455X (gfx1250, wave32, WMMA + async LDS staging)
//
// Sizes (fixed by reference):
//   BATCH=8192, ELSA=64, TEXT=384, HALF=128, COMBINED=256, HIDDEN=16384, K=32
//
// d_out layout (flat concat, return order):
//   [0]                    reconstructed : 8192*256   f32
//   [2,097,152]            h_sparse      : 8192*16384 f32
//   [136,314,880]          combined      : 8192*256   f32
//   [138,412,032]          idx           : 8192*32    (int32 bit pattern)
//
// Workspace layout (byte offsets):
//   0        combined bf16   8192*256*2   = 4 MB
//   4 MB     enc_w   bf16   16384*256*2   = 8 MB
//   12 MB    dec_w^T f32    16384*256*4   = 16 MB
//   28 MB    topk vals f32   8192*32*4    = 1 MB
//   29 MB    topk idx  i32   8192*32*4    = 1 MB
// ---------------------------------------------------------------------------

#define B_ROWS   8192
#define ELSA_D   64
#define TEXT_D   384
#define HALF_D   128
#define COMB_D   256
#define HID_D    16384
#define K_TOP    32

typedef __attribute__((ext_vector_type(16))) __bf16 v16bf;
typedef __attribute__((ext_vector_type(8)))  __bf16 v8bf;
typedef __attribute__((ext_vector_type(8)))  float  v8f;
typedef __attribute__((ext_vector_type(4)))  int    v4i;

// gfx1250 async global->LDS copy path (ASYNCcnt-tracked), with portable fallback.
#if __has_builtin(__builtin_amdgcn_global_load_async_to_lds_b128) && \
    __has_builtin(__builtin_amdgcn_s_wait_asynccnt)
#define USE_ASYNC_LDS 1
typedef __attribute__((address_space(1))) v4i* gbl_v4i_p;
typedef __attribute__((address_space(3))) v4i* lds_v4i_p;
#else
#define USE_ASYNC_LDS 0
#endif

__device__ __forceinline__ unsigned short f2bf(float f) {
  unsigned int u = __float_as_uint(f);
  // round-to-nearest-even bf16 truncation
  unsigned int r = (u + 0x7FFFu + ((u >> 16) & 1u)) >> 16;
  return (unsigned short)r;
}

__device__ __forceinline__ float wsum32(float v) {
#pragma unroll
  for (int off = 16; off > 0; off >>= 1) v += __shfl_xor(v, off, 32);
  return v;
}

__device__ __forceinline__ float gelu_exact(float x) {
  return 0.5f * x * (1.0f + erff(x * 0.70710678118654752f));
}

// ---------------------------------------------------------------------------
// Kernel 1: fused front-end.  One wave32 per row, 8 rows per block.
// elsa MLP -> LN -> GELU -> MLP -> LN;  text likewise;  concat;
// fusion linear -> LN -> L2 normalize.  Emits combined (f32) + bf16 copy.
// ---------------------------------------------------------------------------
__global__ __launch_bounds__(256) void frontend_kernel(
    const float* __restrict__ elsa, const float* __restrict__ text,
    const float* __restrict__ ew1, const float* __restrict__ eb1,
    const float* __restrict__ eg1, const float* __restrict__ ebt1,
    const float* __restrict__ ew2, const float* __restrict__ eb2,
    const float* __restrict__ eg2, const float* __restrict__ ebt2,
    const float* __restrict__ tw1, const float* __restrict__ tb1,
    const float* __restrict__ tg1, const float* __restrict__ tbt1,
    const float* __restrict__ tw2, const float* __restrict__ tb2,
    const float* __restrict__ tg2, const float* __restrict__ tbt2,
    const float* __restrict__ fw,  const float* __restrict__ fb,
    const float* __restrict__ fg,  const float* __restrict__ fbt,
    float* __restrict__ combined_out, unsigned short* __restrict__ combined_bf) {
  const int tid  = threadIdx.x;
  const int lane = tid & 31;
  const int r    = tid >> 5;
  const int row  = blockIdx.x * 8 + r;

  __shared__ float s_x[8][TEXT_D];   // input staging (elsa then text)
  __shared__ float s_v[8][HALF_D];   // post-GELU hidden
  __shared__ float s_c[8][COMB_D];   // e|t concat, then fusion input

  float h[8];

  // ---------------- elsa branch ----------------
  for (int i = lane; i < ELSA_D; i += 32) s_x[r][i] = elsa[row * ELSA_D + i];
  __syncthreads();
#pragma unroll
  for (int j = 0; j < 4; ++j) {
    const int o = j * 32 + lane;
    float acc = eb1[o];
    const float* wr = ew1 + o * ELSA_D;
#pragma unroll 8
    for (int i = 0; i < ELSA_D; ++i) acc = fmaf(wr[i], s_x[r][i], acc);
    h[j] = acc;
  }
  {
    float mean = wsum32(h[0] + h[1] + h[2] + h[3]) * (1.0f / 128.0f);
    float vs = 0.f;
#pragma unroll
    for (int j = 0; j < 4; ++j) { float d = h[j] - mean; vs += d * d; }
    float is = rsqrtf(wsum32(vs) * (1.0f / 128.0f) + 1e-5f);
#pragma unroll
    for (int j = 0; j < 4; ++j) {
      const int o = j * 32 + lane;
      s_v[r][o] = gelu_exact((h[j] - mean) * is * eg1[o] + ebt1[o]);
    }
  }
  __syncthreads();
#pragma unroll
  for (int j = 0; j < 4; ++j) {
    const int o = j * 32 + lane;
    float acc = eb2[o];
    const float* wr = ew2 + o * HALF_D;
#pragma unroll 8
    for (int i = 0; i < HALF_D; ++i) acc = fmaf(wr[i], s_v[r][i], acc);
    h[j] = acc;
  }
  {
    float mean = wsum32(h[0] + h[1] + h[2] + h[3]) * (1.0f / 128.0f);
    float vs = 0.f;
#pragma unroll
    for (int j = 0; j < 4; ++j) { float d = h[j] - mean; vs += d * d; }
    float is = rsqrtf(wsum32(vs) * (1.0f / 128.0f) + 1e-5f);
#pragma unroll
    for (int j = 0; j < 4; ++j) {
      const int o = j * 32 + lane;
      s_c[r][o] = (h[j] - mean) * is * eg2[o] + ebt2[o];
    }
  }
  __syncthreads();

  // ---------------- text branch ----------------
  for (int i = lane; i < TEXT_D; i += 32) s_x[r][i] = text[row * TEXT_D + i];
  __syncthreads();
#pragma unroll
  for (int j = 0; j < 4; ++j) {
    const int o = j * 32 + lane;
    float acc = tb1[o];
    const float* wr = tw1 + o * TEXT_D;
#pragma unroll 8
    for (int i = 0; i < TEXT_D; ++i) acc = fmaf(wr[i], s_x[r][i], acc);
    h[j] = acc;
  }
  {
    float mean = wsum32(h[0] + h[1] + h[2] + h[3]) * (1.0f / 128.0f);
    float vs = 0.f;
#pragma unroll
    for (int j = 0; j < 4; ++j) { float d = h[j] - mean; vs += d * d; }
    float is = rsqrtf(wsum32(vs) * (1.0f / 128.0f) + 1e-5f);
#pragma unroll
    for (int j = 0; j < 4; ++j) {
      const int o = j * 32 + lane;
      s_v[r][o] = gelu_exact((h[j] - mean) * is * tg1[o] + tbt1[o]);
    }
  }
  __syncthreads();
#pragma unroll
  for (int j = 0; j < 4; ++j) {
    const int o = j * 32 + lane;
    float acc = tb2[o];
    const float* wr = tw2 + o * HALF_D;
#pragma unroll 8
    for (int i = 0; i < HALF_D; ++i) acc = fmaf(wr[i], s_v[r][i], acc);
    h[j] = acc;
  }
  {
    float mean = wsum32(h[0] + h[1] + h[2] + h[3]) * (1.0f / 128.0f);
    float vs = 0.f;
#pragma unroll
    for (int j = 0; j < 4; ++j) { float d = h[j] - mean; vs += d * d; }
    float is = rsqrtf(wsum32(vs) * (1.0f / 128.0f) + 1e-5f);
#pragma unroll
    for (int j = 0; j < 4; ++j) {
      const int o = j * 32 + lane;
      s_c[r][HALF_D + o] = (h[j] - mean) * is * tg2[o] + tbt2[o];
    }
  }
  __syncthreads();

  // ---------------- fusion + LN + L2 normalize ----------------
#pragma unroll
  for (int j = 0; j < 8; ++j) {
    const int o = j * 32 + lane;
    float acc = fb[o];
    const float* wr = fw + o * COMB_D;
#pragma unroll 8
    for (int i = 0; i < COMB_D; ++i) acc = fmaf(wr[i], s_c[r][i], acc);
    h[j] = acc;
  }
  {
    float ps = 0.f;
#pragma unroll
    for (int j = 0; j < 8; ++j) ps += h[j];
    float mean = wsum32(ps) * (1.0f / 256.0f);
    float vs = 0.f;
#pragma unroll
    for (int j = 0; j < 8; ++j) { float d = h[j] - mean; vs += d * d; }
    float is = rsqrtf(wsum32(vs) * (1.0f / 256.0f) + 1e-5f);
    float g[8];
    float ss = 0.f;
#pragma unroll
    for (int j = 0; j < 8; ++j) {
      const int o = j * 32 + lane;
      g[j] = (h[j] - mean) * is * fg[o] + fbt[o];
      ss += g[j] * g[j];
    }
    float inv = 1.0f / fmaxf(sqrtf(wsum32(ss)), 1e-12f);
#pragma unroll
    for (int j = 0; j < 8; ++j) {
      const int o = j * 32 + lane;
      const float cv = g[j] * inv;
      combined_out[row * COMB_D + o] = cv;
      combined_bf[row * COMB_D + o] = f2bf(cv);
    }
  }
}

// ---------------------------------------------------------------------------
// Kernel 2: f32 -> bf16 conversion (enc_w), one element per thread.
// ---------------------------------------------------------------------------
__global__ __launch_bounds__(256) void f32_to_bf16_kernel(
    const float* __restrict__ in, unsigned short* __restrict__ out, int n) {
  int i = blockIdx.x * 256 + threadIdx.x;
  if (i < n) out[i] = f2bf(in[i]);
}

// ---------------------------------------------------------------------------
// Kernel 3: transpose dec_w (256 x 16384) -> dec_wT (16384 x 256) via LDS tile.
// ---------------------------------------------------------------------------
__global__ __launch_bounds__(256) void transpose_kernel(
    const float* __restrict__ in, float* __restrict__ out) {
  __shared__ float tile[32][33];
  const int bx = blockIdx.x;          // 512 tiles along hidden
  const int by = blockIdx.y;          // 8 tiles along combined
  const int tx = threadIdx.x & 31;
  const int ty = threadIdx.x >> 5;
#pragma unroll
  for (int s = 0; s < 4; ++s) {
    const int o = by * 32 + ty + s * 8;
    tile[ty + s * 8][tx] = in[(size_t)o * HID_D + bx * 32 + tx];
  }
  __syncthreads();
#pragma unroll
  for (int s = 0; s < 4; ++s) {
    const int hh = bx * 32 + ty + s * 8;
    out[(size_t)hh * COMB_D + by * 32 + tx] = tile[tx][ty + s * 8];
  }
}

// ---------------------------------------------------------------------------
// Kernel 4: encoder GEMM via v_wmma_f32_16x16x32_bf16 with async-LDS B staging.
//   Hpre[m,n] = sum_k A[m,k] * W[n,k] + enc_b[n]      (A:8192x256, W:16384x256)
//
// Block tile = 128(M) x 64(N): the 64-row B strip (full K=256) is staged once
// into LDS via GLOBAL_LOAD_ASYNC_TO_LDS_B128 (ASYNCcnt) and shared by all 8
// waves; each wave computes a 16x64 strip (4 accumulators, K-outer loop, A
// streamed from global and reused across the 4 N-tiles).
//
// LDS B rows padded to 264 bf16 (132 dwords == 4 mod 64 banks) so the 16-lane
// ds_load_b128 B-fragment reads are bank-conflict free.
//
// VGPR layouts per CDNA5 ISA 7.12.2:
//   A 16x32 bf16: lane m=L%16; K-chunks [b..b+7] and [b+16..b+23], b=(L<16?0:8)
//   B 32x16 bf16: lane n=L%16; K-chunks [b..b+7] and [b+16..b+23], b=(L<16?0:8)
//   C/D 16x16 f32: vgpr r, lane L -> D[r + 8*(L/16)][L%16]
// ---------------------------------------------------------------------------
#define B_STRIDE 264   // 256 data + 8 pad (bf16 elements)

__global__ __launch_bounds__(256) void encode_wmma_kernel(
    const unsigned short* __restrict__ Abf,   // [8192][256] bf16
    const unsigned short* __restrict__ Wbf,   // [16384][256] bf16
    const float* __restrict__ enc_b,          // [16384]
    float* __restrict__ Hpre) {               // [8192][16384]
  __shared__ unsigned short sB[64 * B_STRIDE];   // 33 KB

  const int tid   = threadIdx.x;
  const int lane  = tid & 31;
  const int wave  = tid >> 5;
  const int nb    = blockIdx.x & 255;   // 256 N-blocks of 64
  const int mb    = blockIdx.x >> 8;    // 64  M-blocks of 128
  const int n0blk = nb * 64;
  const int m0    = mb * 128 + wave * 16;
  const int lhalf = lane >> 4;
  const int l16   = lane & 15;

  // ---- Stage B strip: 64 rows x 512 B = 2048 x 16 B chunks, 8 per thread ----
  for (int c = tid; c < 64 * 32; c += 256) {
    const int n  = c >> 5;
    const int cc = c & 31;
    const unsigned short* gsrc = Wbf + (size_t)(n0blk + n) * COMB_D + cc * 8;
    unsigned short* ldst = &sB[n * B_STRIDE + cc * 8];
#if USE_ASYNC_LDS
    __builtin_amdgcn_global_load_async_to_lds_b128(
        (gbl_v4i_p)(v4i*)gsrc, (lds_v4i_p)(v4i*)ldst, 0, 0);
#else
    *(uint4*)ldst = *(const uint4*)gsrc;
#endif
  }
#if USE_ASYNC_LDS
  __builtin_amdgcn_s_wait_asynccnt(0);
#endif
  __syncthreads();

  // ---- Compute: K-outer, 4 N-accumulators, A reused across N-tiles ----
  const unsigned short* arow = Abf + (size_t)(m0 + l16) * COMB_D;
  v8f cacc[4];
#pragma unroll
  for (int nt = 0; nt < 4; ++nt) cacc[nt] = {};

#pragma unroll
  for (int kk = 0; kk < 8; ++kk) {
    const int kb = kk * 32 + lhalf * 8;
    v8bf alo = *(const v8bf*)(arow + kb);
    v8bf ahi = *(const v8bf*)(arow + kb + 16);
    v16bf a;
#pragma unroll
    for (int i = 0; i < 8; ++i) { a[i] = alo[i]; a[i + 8] = ahi[i]; }
#pragma unroll
    for (int nt = 0; nt < 4; ++nt) {
      const unsigned short* bp =
          &sB[(nt * 16 + l16) * B_STRIDE + kk * 32 + lhalf * 8];
      v8bf blo = *(const v8bf*)bp;
      v8bf bhi = *(const v8bf*)(bp + 16);
      v16bf b;
#pragma unroll
      for (int i = 0; i < 8; ++i) { b[i] = blo[i]; b[i + 8] = bhi[i]; }
      cacc[nt] = __builtin_amdgcn_wmma_f32_16x16x32_bf16(
          /*neg_a=*/false, a, /*neg_b=*/false, b,
          /*c_mod=*/(short)0, cacc[nt], /*reuse_a=*/false, /*reuse_b=*/false);
    }
  }

#pragma unroll
  for (int nt = 0; nt < 4; ++nt) {
    const int n0 = n0blk + nt * 16;
    const float bias = enc_b[n0 + l16];
    float* out = Hpre + (size_t)(m0 + 8 * lhalf) * HID_D + n0 + l16;
#pragma unroll
    for (int rr = 0; rr < 8; ++rr) out[(size_t)rr * HID_D] = cacc[nt][rr] + bias;
  }
}

// ---------------------------------------------------------------------------
// Kernel 5: per-row top-32 of 16384 (one 256-thread block per row).
// In-place: h_pre (in d_out's h_sparse slot) -> h_sparse.  Each thread owns
// 64 elements + a 64-bit used mask; 32 rounds of block argmax (tie -> lower
// index, matching jax.lax.top_k ordering); only the winning thread rescans.
// ---------------------------------------------------------------------------
__global__ __launch_bounds__(256) void topk_kernel(
    float* __restrict__ H,                 // [8192][16384] in/out
    float* __restrict__ out_vals,          // ws [8192][32] relu'd, descending
    int* __restrict__ out_idx_ws,          // ws [8192][32]
    int* __restrict__ out_idx) {           // d_out idx slot
  const int row  = blockIdx.x;
  const int tid  = threadIdx.x;
  float* rowp = H + (size_t)row * HID_D;
  const int base = tid * 64;

  __shared__ float rv[256];
  __shared__ int   ri[256];
  __shared__ float selv[K_TOP];
  __shared__ int   seli[K_TOP];

  unsigned long long used = 0ull;
  float bv = -3.0e38f; int bi = base;
  for (int e = 0; e < 64; ++e) {
    float v = rowp[base + e];
    if (v > bv) { bv = v; bi = base + e; }
  }

  for (int t = 0; t < K_TOP; ++t) {
    rv[tid] = bv; ri[tid] = bi;
    __syncthreads();
    for (int s = 128; s > 0; s >>= 1) {
      if (tid < s) {
        float ov = rv[tid + s]; int oi = ri[tid + s];
        if (ov > rv[tid] || (ov == rv[tid] && oi < ri[tid])) {
          rv[tid] = ov; ri[tid] = oi;
        }
      }
      __syncthreads();
    }
    const float wv = rv[0]; const int wi = ri[0];
    if (tid == 0) { selv[t] = wv; seli[t] = wi; }
    if (wi >= base && wi < base + 64) {
      used |= 1ull << (wi - base);
      bv = -3.0e38f; bi = base;
      for (int e = 0; e < 64; ++e) {
        if ((used >> e) & 1ull) continue;
        float v = rowp[base + e];
        if (v > bv) { bv = v; bi = base + e; }
      }
    }
    __syncthreads();
  }

  // Zero the row, then scatter relu(top-k).
  float4 z = make_float4(0.f, 0.f, 0.f, 0.f);
  float4* rp4 = (float4*)rowp;
  for (int i = tid; i < HID_D / 4; i += 256) rp4[i] = z;
  __syncthreads();
  if (tid < K_TOP) {
    const float v  = selv[tid];
    const int   ix = seli[tid];
    const float rl = v > 0.f ? v : 0.f;
    rowp[ix] = rl;
    out_vals[row * K_TOP + tid]   = rl;
    out_idx_ws[row * K_TOP + tid] = ix;
    out_idx[row * K_TOP + tid]    = ix;   // int32 bit pattern in output slot
  }
}

// ---------------------------------------------------------------------------
// Kernel 6: sparse decode.  recon[row, o] = dec_b[o] + sum_j v_j * dec_wT[i_j][o]
// One 256-thread block per row; dec_wT rows contiguous -> coalesced.
// ---------------------------------------------------------------------------
__global__ __launch_bounds__(256) void decode_kernel(
    const float* __restrict__ vals, const int* __restrict__ idxs,
    const float* __restrict__ decT, const float* __restrict__ dec_b,
    float* __restrict__ recon) {
  const int row = blockIdx.x;
  const int tid = threadIdx.x;
  __shared__ float sv[K_TOP];
  __shared__ int   si[K_TOP];
  if (tid < K_TOP) { sv[tid] = vals[row * K_TOP + tid]; si[tid] = idxs[row * K_TOP + tid]; }
  __syncthreads();
  float acc = dec_b[tid];
#pragma unroll 4
  for (int j = 0; j < K_TOP; ++j)
    acc = fmaf(sv[j], decT[(size_t)si[j] * COMB_D + tid], acc);
  recon[row * COMB_D + tid] = acc;
}

// ---------------------------------------------------------------------------
extern "C" void kernel_launch(void* const* d_in, const int* in_sizes, int n_in,
                              void* d_out, int out_size, void* d_ws, size_t ws_size,
                              hipStream_t stream) {
  (void)in_sizes; (void)n_in; (void)out_size; (void)ws_size;
  const float* elsa  = (const float*)d_in[0];
  const float* text  = (const float*)d_in[1];
  // d_in[2] = k (fixed at 32 per reference setup)
  const float* ew1 = (const float*)d_in[3];  const float* eb1 = (const float*)d_in[4];
  const float* eg1 = (const float*)d_in[5];  const float* ebt1= (const float*)d_in[6];
  const float* ew2 = (const float*)d_in[7];  const float* eb2 = (const float*)d_in[8];
  const float* eg2 = (const float*)d_in[9];  const float* ebt2= (const float*)d_in[10];
  const float* tw1 = (const float*)d_in[11]; const float* tb1 = (const float*)d_in[12];
  const float* tg1 = (const float*)d_in[13]; const float* tbt1= (const float*)d_in[14];
  const float* tw2 = (const float*)d_in[15]; const float* tb2 = (const float*)d_in[16];
  const float* tg2 = (const float*)d_in[17]; const float* tbt2= (const float*)d_in[18];
  const float* fw  = (const float*)d_in[19]; const float* fb  = (const float*)d_in[20];
  const float* fg  = (const float*)d_in[21]; const float* fbt = (const float*)d_in[22];
  const float* enc_w = (const float*)d_in[23]; const float* enc_b = (const float*)d_in[24];
  const float* dec_w = (const float*)d_in[25]; const float* dec_b = (const float*)d_in[26];

  float* recon = (float*)d_out;
  float* hsp   = recon + (size_t)B_ROWS * COMB_D;
  float* comb  = hsp + (size_t)B_ROWS * HID_D;
  int*   idxo  = (int*)(comb + (size_t)B_ROWS * COMB_D);

  char* ws = (char*)d_ws;
  unsigned short* comb_bf = (unsigned short*)(ws);
  unsigned short* encw_bf = (unsigned short*)(ws + ((size_t)4 << 20));
  float* decT  = (float*)(ws + ((size_t)12 << 20));
  float* tvals = (float*)(ws + ((size_t)28 << 20));
  int*   tidx  = (int*)  (ws + ((size_t)29 << 20));

  frontend_kernel<<<B_ROWS / 8, 256, 0, stream>>>(
      elsa, text, ew1, eb1, eg1, ebt1, ew2, eb2, eg2, ebt2,
      tw1, tb1, tg1, tbt1, tw2, tb2, tg2, tbt2, fw, fb, fg, fbt,
      comb, comb_bf);

  f32_to_bf16_kernel<<<(HID_D * COMB_D) / 256, 256, 0, stream>>>(
      enc_w, encw_bf, HID_D * COMB_D);

  transpose_kernel<<<dim3(HID_D / 32, COMB_D / 32), 256, 0, stream>>>(dec_w, decT);

  // 64 M-blocks (128 rows) x 256 N-blocks (64 cols) = 16384 blocks.
  encode_wmma_kernel<<<16384, 256, 0, stream>>>(comb_bf, encw_bf, enc_b, hsp);

  topk_kernel<<<B_ROWS, 256, 0, stream>>>(hsp, tvals, tidx, idxo);

  decode_kernel<<<B_ROWS, 256, 0, stream>>>(tvals, tidx, decT, dec_b, recon);
}
